// GridNetBlock_14946486190410
// MI455X (gfx1250) — matrix operand order repaired and checked by reading, not compiled, not run on hardware
//
#include <hip/hip_runtime.h>
#include <hip/hip_bf16.h>

typedef __attribute__((ext_vector_type(16))) _Float16 v16h;
typedef __attribute__((ext_vector_type(8)))  _Float16 v8h;
typedef __attribute__((ext_vector_type(8)))  float    v8f;
typedef _Float16 half_t;

// Problem constants (from reference)
static constexpr int Bb   = 2;
static constexpr int Cc   = 48;
static constexpr int Tt   = 250;
static constexpr int Qq   = 65;
static constexpr int NROW = Bb * Tt * Qq;      // 32500 (= 500*65 = 130*250)
static constexpr int NA   = Bb * Tt;           // 500 intra seq batch
static constexpr int LA   = Qq - 3;            // 62  intra unfold len
static constexpr int NB   = Bb * Qq;           // 130 inter seq batch
static constexpr int LB   = Tt - 3;            // 247 inter unfold len
static constexpr int UROW = NB * LB;           // 32110 >= NA*LA
static constexpr int HID  = 192;               // LSTM hidden
static constexpr int G4   = 4 * HID;           // 768 gates
static constexpr int KC   = 2 * HID;           // 384 fused K (h | u)
static constexpr float EPSV = 1e-5f;

__device__ __forceinline__ float sigf(float x) { return 1.f / (1.f + __expf(-x)); }

// Load 16 contiguous halves (two 16B chunks) into a v16h, no bounds checks.
__device__ __forceinline__ v16h ld16(const half_t* p)
{
  v8h a0, a1;
  __builtin_memcpy(&a0, p, 16);
  __builtin_memcpy(&a1, p + 8, 16);
  return __builtin_shufflevector(a0, a1, 0, 1, 2, 3, 4, 5, 6, 7,
                                 8, 9, 10, 11, 12, 13, 14, 15);
}
// Load A-style fragment: halves 0..7 at p, halves 8..15 at p+16 (K split +16).
__device__ __forceinline__ v16h ld16a(const half_t* p)
{
  v8h a0, a1;
  __builtin_memcpy(&a0, p, 16);
  __builtin_memcpy(&a1, p + 16, 16);
  return __builtin_shufflevector(a0, a1, 0, 1, 2, 3, 4, 5, 6, 7,
                                 8, 9, 10, 11, 12, 13, 14, 15);
}

// ---------------------------------------------------------------------------
// Generic WMMA GEMM:  C[M,N] = A[M,K] * op(B) (+bias[N])
//   A: f16 row-major MxK.   NT: B f16 row-major NxK.   NN: B f16 row-major KxN
//   Output f32 (Cf) or f16 (Ch) — exactly one non-null.
// Wave = one 16x16 tile; 8 waves/block. Rows/cols are CLAMPED (WMMA row r only
// feeds C row r, so duplicates are harmless and never stored); only the final
// partial K-chunk needs zero-fill.
// ---------------------------------------------------------------------------
template <bool NT>
__global__ void __launch_bounds__(256)
k_gemm(const half_t* __restrict__ A, const half_t* __restrict__ Bm,
       const float* __restrict__ bias, float* __restrict__ Cf,
       half_t* __restrict__ Ch, int M, int N, int K)
{
  int M16 = (M + 15) >> 4, N16 = (N + 15) >> 4;
  int wave = threadIdx.x >> 5;
  int lane = threadIdx.x & 31;
  long tile = (long)blockIdx.x * 8 + wave;
  if (tile >= (long)M16 * N16) return;
  int tm = (int)(tile / N16), tn = (int)(tile % N16);
  int row0 = tm << 4, col0 = tn << 4;

  int aSel = (lane & 16) ? 16 : 0;
  int bSel = (lane & 16) ? 16 : 0;
  int rA   = min(row0 + (lane & 15), M - 1);   // clamped A row
  int bn   = min(col0 + (lane & 15), N - 1);   // clamped B row/col

  const half_t* aRow = A + (size_t)rA * K;
  const half_t* bRow = NT ? (Bm + (size_t)bn * K) : (Bm + bn);

  int kFull = K & ~31;
  v8f acc = {};
  for (int k0 = 0; k0 < kFull; k0 += 32) {
    v16h a = ld16(aRow + k0 + aSel);
    v16h b;
    if (NT) {
      b = ld16(bRow + k0 + bSel);
    } else {
#pragma unroll
      for (int h = 0; h < 16; ++h)
        b[h] = bRow[(size_t)(k0 + bSel + h) * N];
    }
    acc = __builtin_amdgcn_wmma_f32_16x16x32_f16(false, a, false, b,
                                                 (short)0, acc, false, false);
  }
  if (kFull < K) {
    v16h a, b;
#pragma unroll
    for (int h = 0; h < 16; ++h) {
      int k = kFull + aSel + h;
      a[h] = (k < K) ? aRow[k] : (half_t)0;
    }
#pragma unroll
    for (int h = 0; h < 16; ++h) {
      int k = kFull + bSel + h;
      b[h] = (k < K) ? (NT ? bRow[k] : bRow[(size_t)k * N]) : (half_t)0;
    }
    acc = __builtin_amdgcn_wmma_f32_16x16x32_f16(false, a, false, b,
                                                 (short)0, acc, false, false);
  }

  int col = col0 + (lane & 15);
  if (col >= N) return;
  float bb = bias ? bias[col] : 0.f;
  int rb = row0 + ((lane & 16) ? 8 : 0);
  if (Ch) {
#pragma unroll
    for (int j = 0; j < 8; ++j) {
      int rr = rb + j;
      if (rr < M) Ch[(size_t)rr * N + col] = (half_t)(acc[j] + bb);
    }
  } else {
#pragma unroll
    for (int j = 0; j < 8; ++j) {
      int rr = rb + j;
      if (rr < M) Cf[(size_t)rr * N + col] = acc[j] + bb;
    }
  }
}

// ---------------------------------------------------------------------------
// Fused BiLSTM direction scan. Block owns 16 batch rows for all L steps.
// gates(16x768) = [h_prev(16x192) | u_t(16x192)] x Wc^T(768x384) via WMMA.
// K chunks 0..5 read h from LDS (ds_load_b128), 6..11 read u from global
// (clamped batch row — out-of-range rows are computed but never stored).
// ---------------------------------------------------------------------------
__global__ void __launch_bounds__(256)
k_lstm_scan(const half_t* __restrict__ u, const half_t* __restrict__ Wc,
            const float* __restrict__ bi, const float* __restrict__ bh,
            half_t* __restrict__ ycat, int Nb, int L, int reverse, int dirOff)
{
  __shared__ __align__(16) half_t h_sh[16 * HID];   //  6 KB
  __shared__ float g_sh[16 * G4];                   // 48 KB
  int n0   = blockIdx.x * 16;
  int tid  = threadIdx.x;
  int wave = tid >> 5, lane = tid & 31;
  int r    = lane & 15;
  int aSel = (lane & 16) ? 16 : 0;
  int bSel = (lane & 16) ? 16 : 0;
  int grc  = min(n0 + r, Nb - 1);                   // clamped batch row for u

  float c_reg[12];
#pragma unroll
  for (int j = 0; j < 12; ++j) c_reg[j] = 0.f;
  for (int i = tid; i < 16 * HID; i += 256) h_sh[i] = (half_t)0;
  __syncthreads();

  for (int step = 0; step < L; ++step) {
    int l = reverse ? (L - 1 - step) : step;
    const half_t* uRow = u + ((size_t)grc * L + l) * HID;
    const half_t* hRow = &h_sh[r * HID];
    // ---- WMMA phase: each wave computes 6 of the 48 N-tiles ----
    for (int t6 = 0; t6 < 6; ++t6) {
      int tn = wave * 6 + t6;
      const half_t* wRow = Wc + (size_t)(tn * 16 + r) * KC;
      v8f acc = {};
#pragma unroll
      for (int kc = 0; kc < 6; ++kc) {              // h region (LDS)
        v16h a = ld16(hRow + kc * 32 + aSel);
        v16h b = ld16(wRow + kc * 32 + bSel);
        acc = __builtin_amdgcn_wmma_f32_16x16x32_f16(false, a, false, b,
                                                     (short)0, acc, false, false);
      }
#pragma unroll
      for (int kc = 6; kc < 12; ++kc) {             // u region (global)
        v16h a = ld16(uRow + (kc - 6) * 32 + aSel);
        v16h b = ld16(wRow + kc * 32 + bSel);
        acc = __builtin_amdgcn_wmma_f32_16x16x32_f16(false, a, false, b,
                                                     (short)0, acc, false, false);
      }
      int col = tn * 16 + r;
      int rb  = (lane & 16) ? 8 : 0;
      float bbias = bi[col] + bh[col];
#pragma unroll
      for (int j = 0; j < 8; ++j) g_sh[(rb + j) * G4 + col] = acc[j] + bbias;
    }
    __syncthreads();
    // ---- activation phase: 12 (row,hid) cells per thread ----
#pragma unroll
    for (int j = 0; j < 12; ++j) {
      int cell = tid * 12 + j;
      int rr = cell / HID, hid = cell % HID;
      float gi = g_sh[rr * G4 + hid];
      float gf = g_sh[rr * G4 + HID + hid];
      float gg = g_sh[rr * G4 + 2 * HID + hid];
      float go = g_sh[rr * G4 + 3 * HID + hid];
      float cnew = sigf(gf) * c_reg[j] + sigf(gi) * tanhf(gg);
      float hnew = sigf(go) * tanhf(cnew);
      c_reg[j] = cnew;
      h_sh[rr * HID + hid] = (half_t)hnew;
      int gr = n0 + rr;
      if (gr < Nb)
        ycat[((size_t)gr * L + l) * (2 * HID) + dirOff + hid] = (half_t)hnew;
    }
    __syncthreads();
  }
}

// ---------------------------------------------------------------------------
// Small elementwise / reduction kernels
// ---------------------------------------------------------------------------
__global__ void k_f2h(const float* __restrict__ s, half_t* __restrict__ d, long n)
{
  long i = (long)blockIdx.x * blockDim.x + threadIdx.x;
  if (i < n) d[i] = (half_t)s[i];
}

__global__ void k_build_wc(const float* __restrict__ wi, const float* __restrict__ wh,
                           half_t* __restrict__ wc)
{
  long i = (long)blockIdx.x * blockDim.x + threadIdx.x;
  if (i >= (long)G4 * KC) return;
  int g = (int)(i / KC), j = (int)(i % KC);
  wc[i] = (half_t)((j < HID) ? wh[(size_t)g * HID + j] : wi[(size_t)g * HID + (j - HID)]);
}

// channel LN (ln_clamp over C) -> seq layout. mode 0: row=(b*T+t)*Q+q ; mode 1: row=(b*Q+q)*T+t
__global__ void k_ln_ch(const float* __restrict__ x, const float* __restrict__ g,
                        const float* __restrict__ bta, float* __restrict__ seq,
                        half_t* __restrict__ seqh, int mode)
{
  int idx = blockIdx.x * blockDim.x + threadIdx.x;
  if (idx >= NROW) return;
  int b = idx / (Tt * Qq);
  int rem = idx % (Tt * Qq);
  int t = rem / Qq, q = rem % Qq;
  float s = 0.f, s2 = 0.f;
  for (int c = 0; c < Cc; ++c) {
    float v = x[(((size_t)b * Cc + c) * Tt + t) * Qq + q];
    s += v; s2 += v * v;
  }
  float mu = s / Cc;
  float var = s2 / Cc - mu * mu;
  float rs = rsqrtf(fmaxf(var, EPSV));
  size_t row = (mode == 0) ? ((size_t)(b * Tt + t) * Qq + q)
                           : ((size_t)(b * Qq + q) * Tt + t);
  for (int c = 0; c < Cc; ++c) {
    float v = x[(((size_t)b * Cc + c) * Tt + t) * Qq + q];
    float o = (v - mu) * rs * g[c] + bta[c];
    seq[row * Cc + c]  = o;
    seqh[row * Cc + c] = (half_t)o;
  }
}

// 4-head MHA (dh=12) with online softmax; qkv rows (n,l) of 144; out f16 rows of 48
__global__ void k_mha(const float* __restrict__ qkv, half_t* __restrict__ o16,
                      int L, float scale)
{
  int row = blockIdx.x * blockDim.x + threadIdx.x;
  if (row >= NROW) return;
  int n = row / L;
  size_t nb = (size_t)n * L * 144;
  for (int h = 0; h < 4; ++h) {
    float q[12], o[12];
#pragma unroll
    for (int e = 0; e < 12; ++e) {
      q[e] = qkv[(size_t)row * 144 + h * 12 + e];
      o[e] = 0.f;
    }
    float mx = -1e30f, den = 0.f;
    for (int m = 0; m < L; ++m) {
      const float* kr = &qkv[nb + (size_t)m * 144 + 48 + h * 12];
      float sd = 0.f;
#pragma unroll
      for (int e = 0; e < 12; ++e) sd += q[e] * kr[e];
      sd *= scale;
      float nm = fmaxf(mx, sd);
      float corr = __expf(mx - nm);
      float w = __expf(sd - nm);
      den = den * corr + w;
      const float* vr = &qkv[nb + (size_t)m * 144 + 96 + h * 12];
#pragma unroll
      for (int e = 0; e < 12; ++e) o[e] = o[e] * corr + w * vr[e];
      mx = nm;
    }
    float inv = 1.f / den;
#pragma unroll
    for (int e = 0; e < 12; ++e)
      o16[(size_t)row * 48 + h * 12 + e] = (half_t)(o[e] * inv);
  }
}

// out = torch_ln(a + b) over last dim 48
__global__ void k_add_ln(const float* __restrict__ a, const float* __restrict__ b,
                         const float* __restrict__ g, const float* __restrict__ beta,
                         float* __restrict__ out)
{
  int row = blockIdx.x * blockDim.x + threadIdx.x;
  if (row >= NROW) return;
  float v[48];
  float s = 0.f, s2 = 0.f;
  for (int c = 0; c < 48; ++c) {
    float x = a[(size_t)row * 48 + c] + b[(size_t)row * 48 + c];
    v[c] = x; s += x; s2 += x * x;
  }
  float mu = s / 48.f;
  float var = s2 / 48.f - mu * mu;
  float rs = rsqrtf(var + EPSV);
  for (int c = 0; c < 48; ++c)
    out[(size_t)row * 48 + c] = (v[c] - mu) * rs * g[c] + beta[c];
}

// unfold: u[(n*L+l)*192 + c*4+k] = seq2[(n*Qlen + l+k)*48 + c]  (f16)
__global__ void k_unfold(const float* __restrict__ seq2, half_t* __restrict__ u,
                         int N, int Qlen, int L)
{
  long idx = (long)blockIdx.x * blockDim.x + threadIdx.x;
  if (idx >= (long)N * L * HID) return;
  int ck = (int)(idx % HID);
  long nl = idx / HID;
  int l = (int)(nl % L), n = (int)(nl / L);
  int c = ck >> 2, k = ck & 3;
  u[idx] = (half_t)seq2[((size_t)n * Qlen + l + k) * 48 + c];
}

// conv-transpose scatter + residual. mode 0 (intra): n=b*T+t, pos=q ; mode 1 (inter): n=b*Q+q, pos=t
__global__ void k_convtr_add(const float* __restrict__ y2, const float* __restrict__ bias,
                             const float* __restrict__ resid, float* __restrict__ outp,
                             int L, int mode)
{
  int idx = blockIdx.x * blockDim.x + threadIdx.x;
  if (idx >= Bb * Cc * Tt * Qq) return;
  int q = idx % Qq;
  int t = (idx / Qq) % Tt;
  int c = (idx / (Qq * Tt)) % Cc;
  int b = idx / (Qq * Tt * Cc);
  int n, pos;
  if (mode == 0) { n = b * Tt + t; pos = q; }
  else           { n = b * Qq + q; pos = t; }
  float s = bias[c];
#pragma unroll
  for (int k = 0; k < 4; ++k) {
    int l = pos - k;
    if (l >= 0 && l < L) s += y2[((size_t)n * L + l) * HID + c * 4 + k];
  }
  outp[idx] = resid[idx] + s;
}

// (b,c,t,f) -> f16 [(b*T+t)*F+f][c]
__global__ void k_bt16(const float* __restrict__ inp, half_t* __restrict__ o)
{
  int idx = blockIdx.x * blockDim.x + threadIdx.x;
  if (idx >= Bb * Cc * Tt * Qq) return;
  int c = idx % 48;
  int f = (idx / 48) % 65;
  int t = (idx / (48 * 65)) % 250;
  int b = idx / (48 * 65 * 250);
  o[idx] = (half_t)inp[(((size_t)b * 48 + c) * 250 + t) * 65 + f];
}

// head projection LN: bias+PReLU then ln_clamp over (e,f); out f16 (h,b,t, e*65+f)
__global__ void k_head_ln(const float* __restrict__ Y, const float* __restrict__ bias,
                          const float* __restrict__ a, const float* __restrict__ g,
                          const float* __restrict__ beta, half_t* __restrict__ out, int dim)
{
  int blk = blockIdx.x;
  int h = blk / (Bb * Tt);
  int rem = blk % (Bb * Tt);
  int b = rem / Tt, t = rem % Tt;
  int NE = 4 * dim;
  int n = dim * 65;
  int tid = threadIdx.x;
  __shared__ float red[256];
  float al = a[h];
  float s = 0.f, s2 = 0.f;
  for (int i = tid; i < n; i += 128) {
    int e = i / 65, f = i % 65;
    float v = Y[(((size_t)(b * Tt + t)) * 65 + f) * NE + h * dim + e] + bias[h * dim + e];
    v = v >= 0.f ? v : al * v;
    s += v; s2 += v * v;
  }
  red[tid] = s; red[128 + tid] = s2;
  __syncthreads();
  for (int off = 64; off > 0; off >>= 1) {
    if (tid < off) { red[tid] += red[tid + off]; red[128 + tid] += red[128 + tid + off]; }
    __syncthreads();
  }
  float mu = red[0] / n;
  float var = red[128] / n - mu * mu;
  float rs = rsqrtf(fmaxf(var, EPSV));
  __syncthreads();
  for (int i = tid; i < n; i += 128) {
    int e = i / 65, f = i % 65;
    float v = Y[(((size_t)(b * Tt + t)) * 65 + f) * NE + h * dim + e] + bias[h * dim + e];
    v = v >= 0.f ? v : al * v;
    float o = (v - mu) * rs * g[(h * dim + e) * 65 + f] + beta[(h * dim + e) * 65 + f];
    out[((size_t)(h * Bb + b) * Tt + t) * n + e * 65 + f] = (half_t)o;
  }
}

// row softmax (scaled) -> f16 probs
__global__ void k_softmax(const float* __restrict__ S, half_t* __restrict__ P,
                          int rows, int n, float scale)
{
  int r = blockIdx.x * blockDim.x + threadIdx.x;
  if (r >= rows) return;
  const float* sr = S + (size_t)r * n;
  half_t* pr = P + (size_t)r * n;
  float mx = -1e30f;
  for (int m = 0; m < n; ++m) mx = fmaxf(mx, sr[m] * scale);
  float den = 0.f;
  for (int m = 0; m < n; ++m) den += __expf(sr[m] * scale - mx);
  float inv = 1.f / den;
  for (int m = 0; m < n; ++m) pr[m] = (half_t)(__expf(sr[m] * scale - mx) * inv);
}

// Vo (h,b,t, d*65+f) -> f16 A-layout [(b*T+t)*F+f][h*12+d]
__global__ void k_vo16(const float* __restrict__ vo, half_t* __restrict__ o)
{
  int idx = blockIdx.x * blockDim.x + threadIdx.x;
  if (idx >= Bb * 48 * Tt * Qq) return;
  int c = idx % 48;
  int f = (idx / 48) % 65;
  int t = (idx / (48 * 65)) % 250;
  int b = idx / (48 * 65 * 250);
  int h = c / 12, d = c % 12;
  o[idx] = (half_t)vo[((size_t)(h * Bb + b) * 250 + t) * 780 + d * 65 + f];
}

// PReLU + ln_clamp over (c,f) per (b,t); add inter residual; write f32 (b,c,t,f) and f16 [(b,t,f),c]
__global__ void k_pj_ln_add(const float* __restrict__ pj, const float* __restrict__ pa,
                            const float* __restrict__ pg, const float* __restrict__ pbt,
                            const float* __restrict__ interb, float* __restrict__ outsum,
                            half_t* __restrict__ out16)
{
  int blk = blockIdx.x;
  int b = blk / Tt, t = blk % Tt;
  int tid = threadIdx.x;
  __shared__ float red[512];
  const int n = 48 * 65;
  float al = pa[0];
  float s = 0.f, s2 = 0.f;
  for (int i = tid; i < n; i += 256) {
    int c = i / 65, f = i % 65;
    float v = pj[(((size_t)(b * Tt + t)) * 65 + f) * 48 + c];
    v = v >= 0.f ? v : al * v;
    s += v; s2 += v * v;
  }
  red[tid] = s; red[256 + tid] = s2;
  __syncthreads();
  for (int off = 128; off > 0; off >>= 1) {
    if (tid < off) { red[tid] += red[tid + off]; red[256 + tid] += red[256 + tid + off]; }
    __syncthreads();
  }
  float mu = red[0] / n;
  float var = red[256] / n - mu * mu;
  float rs = rsqrtf(fmaxf(var, EPSV));
  __syncthreads();
  for (int i = tid; i < n; i += 256) {
    int c = i / 65, f = i % 65;
    size_t pidx = (((size_t)(b * Tt + t)) * 65 + f) * 48 + c;
    float v = pj[pidx];
    v = v >= 0.f ? v : al * v;
    float o = (v - mu) * rs * pg[c * 65 + f] + pbt[c * 65 + f];
    size_t bidx = (((size_t)b * 48 + c) * 250 + t) * 65 + f;
    float ov = o + interb[bidx];
    outsum[bidx] = ov;
    out16[pidx] = (half_t)ov;
  }
}

// depthwise 3x3 conv (pad 1) on pi (b,t,f,384) + GLU with exact GELU -> f16 [(b,t,f),192]
__global__ void k_dwglu(const half_t* __restrict__ pi, const float* __restrict__ w,
                        const float* __restrict__ bconv, half_t* __restrict__ g16)
{
  long idx = (long)blockIdx.x * blockDim.x + threadIdx.x;
  if (idx >= (long)Bb * Tt * Qq * 192) return;
  int m = (int)(idx % 192);
  long r = idx / 192;
  int f = (int)(r % 65);
  long r2 = r / 65;
  int t = (int)(r2 % 250);
  int b = (int)(r2 / 250);
  float x1 = bconv[m], x2 = bconv[192 + m];
#pragma unroll
  for (int dt = 0; dt < 3; ++dt) {
    int tt = t + dt - 1;
    if (tt < 0 || tt >= 250) continue;
#pragma unroll
    for (int df = 0; df < 3; ++df) {
      int ff = f + df - 1;
      if (ff < 0 || ff >= 65) continue;
      size_t base = (((size_t)(b * 250 + tt)) * 65 + ff) * 384;
      x1 += (float)pi[base + m]       * w[m * 9 + dt * 3 + df];
      x2 += (float)pi[base + 192 + m] * w[(192 + m) * 9 + dt * 3 + df];
    }
  }
  float ge = 0.5f * x1 * (1.f + erff(x1 * 0.70710678f));
  g16[idx] = (half_t)(ge * x2);
}

__global__ void k_final(const float* __restrict__ ffn, const float* __restrict__ outsum,
                        float* __restrict__ dout)
{
  int idx = blockIdx.x * blockDim.x + threadIdx.x;
  if (idx >= Bb * Cc * Tt * Qq) return;
  int f = idx % 65;
  int t = (idx / 65) % 250;
  int c = (idx / (65 * 250)) % 48;
  int b = idx / (65 * 250 * 48);
  dout[idx] = outsum[idx] + ffn[(((size_t)(b * 250 + t)) * 65 + f) * 48 + c];
}

// ---------------------------------------------------------------------------
// Host orchestration
// ---------------------------------------------------------------------------
static inline void gemm(hipStream_t s, const half_t* A, const half_t* Bm,
                        const float* bias, float* Cf, half_t* Ch,
                        int M, int N, int K, bool nt)
{
  long tiles = (long)((M + 15) / 16) * ((N + 15) / 16);
  int blocks = (int)((tiles + 7) / 8);
  if (nt) k_gemm<true ><<<blocks, 256, 0, s>>>(A, Bm, bias, Cf, Ch, M, N, K);
  else    k_gemm<false><<<blocks, 256, 0, s>>>(A, Bm, bias, Cf, Ch, M, N, K);
}

static inline int g256(long n) { return (int)((n + 255) / 256); }

extern "C" void kernel_launch(void* const* d_in, const int* in_sizes, int n_in,
                              void* d_out, int out_size, void* d_ws, size_t ws_size,
                              hipStream_t stream)
{
  (void)in_sizes; (void)n_in; (void)out_size; (void)ws_size;
  const float* x         = (const float*)d_in[0];
  const float* af_in_w   = (const float*)d_in[1];
  const float* af_in_b   = (const float*)d_in[2];
  const float* af_out_w  = (const float*)d_in[3];
  const float* af_out_b  = (const float*)d_in[4];
  const float* at_in_w   = (const float*)d_in[5];
  const float* at_in_b   = (const float*)d_in[6];
  const float* at_out_w  = (const float*)d_in[7];
  const float* at_out_b  = (const float*)d_in[8];
  const float* naf_g     = (const float*)d_in[9];
  const float* naf_b     = (const float*)d_in[10];
  const float* nat_g     = (const float*)d_in[11];
  const float* nat_b     = (const float*)d_in[12];
  const float* intra_ng  = (const float*)d_in[13];
  const float* intra_nb  = (const float*)d_in[14];
  const float* inter_ng  = (const float*)d_in[15];
  const float* inter_nb  = (const float*)d_in[16];
  const float* intra_wif = (const float*)d_in[17];
  const float* intra_whf = (const float*)d_in[18];
  const float* intra_bif = (const float*)d_in[19];
  const float* intra_bhf = (const float*)d_in[20];
  const float* intra_wib = (const float*)d_in[21];
  const float* intra_whb = (const float*)d_in[22];
  const float* intra_bib = (const float*)d_in[23];
  const float* intra_bhb = (const float*)d_in[24];
  const float* inter_wif = (const float*)d_in[25];
  const float* inter_whf = (const float*)d_in[26];
  const float* inter_bif = (const float*)d_in[27];
  const float* inter_bhf = (const float*)d_in[28];
  const float* inter_wib = (const float*)d_in[29];
  const float* inter_whb = (const float*)d_in[30];
  const float* inter_bib = (const float*)d_in[31];
  const float* inter_bhb = (const float*)d_in[32];
  const float* il_w      = (const float*)d_in[33];
  const float* il_b      = (const float*)d_in[34];
  const float* itl_w     = (const float*)d_in[35];
  const float* itl_b     = (const float*)d_in[36];
  const float* qw        = (const float*)d_in[37];
  const float* qb        = (const float*)d_in[38];
  const float* qa        = (const float*)d_in[39];
  const float* qg        = (const float*)d_in[40];
  const float* qbt       = (const float*)d_in[41];
  const float* kw        = (const float*)d_in[42];
  const float* kb        = (const float*)d_in[43];
  const float* ka        = (const float*)d_in[44];
  const float* kg        = (const float*)d_in[45];
  const float* kbt       = (const float*)d_in[46];
  const float* vw        = (const float*)d_in[47];
  const float* vb        = (const float*)d_in[48];
  const float* va        = (const float*)d_in[49];
  const float* vg        = (const float*)d_in[50];
  const float* vbt       = (const float*)d_in[51];
  const float* pw        = (const float*)d_in[52];
  const float* pb        = (const float*)d_in[53];
  const float* pa        = (const float*)d_in[54];
  const float* pg        = (const float*)d_in[55];
  const float* pbt       = (const float*)d_in[56];
  const float* fin_w     = (const float*)d_in[57];
  const float* fin_b     = (const float*)d_in[58];
  const float* fdw_w     = (const float*)d_in[59];
  const float* fdw_b     = (const float*)d_in[60];
  const float* fout_w    = (const float*)d_in[61];
  const float* fout_b    = (const float*)d_in[62];

  // ---- workspace arena (~230 MB) ----
  char* base = (char*)d_ws;
  size_t off = 0;
  auto af = [&](size_t n) -> float* {
    off = (off + 255) & ~(size_t)255; float* p = (float*)(base + off); off += n * 4; return p;
  };
  auto ah = [&](size_t n) -> half_t* {
    off = (off + 255) & ~(size_t)255; half_t* p = (half_t*)(base + off); off += n * 2; return p;
  };

  half_t* wc_if  = ah((size_t)G4 * KC);
  half_t* wc_ib  = ah((size_t)G4 * KC);
  half_t* wc_ef  = ah((size_t)G4 * KC);
  half_t* wc_eb  = ah((size_t)G4 * KC);
  half_t* ilw16  = ah((size_t)KC * HID);
  half_t* itlw16 = ah((size_t)KC * HID);
  half_t* afin16 = ah(144 * 48);
  half_t* afout16= ah(48 * 48);
  half_t* atin16 = ah(144 * 48);
  half_t* atout16= ah(48 * 48);
  half_t* qw16   = ah(32 * 48);
  half_t* kw16   = ah(32 * 48);
  half_t* vw16   = ah(48 * 48);
  half_t* pw16   = ah(48 * 48);
  half_t* finw16 = ah(384 * 48);
  half_t* foutw16= ah(48 * 192);

  float*  seq    = af((size_t)NROW * 48);
  half_t* seqh   = ah((size_t)NROW * 48);
  float*  qkv    = af((size_t)NROW * 144);
  half_t* o16    = ah((size_t)NROW * 48);
  float*  att    = af((size_t)NROW * 48);
  float*  seq2   = af((size_t)NROW * 48);
  half_t* u16    = ah((size_t)UROW * HID);
  half_t* rcat   = ah((size_t)UROW * 2 * HID);
  float*  y2     = af((size_t)UROW * HID);
  float*  intra  = af((size_t)Bb * Cc * Tt * Qq);
  float*  inter  = af((size_t)Bb * Cc * Tt * Qq);
  half_t* bt16   = ah((size_t)NROW * 48);
  float*  Qy     = af((size_t)NROW * 32);
  float*  Ky     = af((size_t)NROW * 32);
  float*  Vy     = af((size_t)NROW * 48);
  half_t* Qf     = ah((size_t)4 * 2 * 250 * 520);
  half_t* Kf     = ah((size_t)4 * 2 * 250 * 520);
  half_t* Vf     = ah((size_t)4 * 2 * 250 * 780);
  float*  sc     = af((size_t)8 * 250 * 250);
  half_t* pm     = ah((size_t)8 * 250 * 250);
  float*  vo     = af((size_t)8 * 250 * 780);
  half_t* vo16   = ah((size_t)NROW * 48);
  float*  pj     = af((size_t)NROW * 48);
  float*  outsum = af((size_t)Bb * Cc * Tt * Qq);
  half_t* out16  = ah((size_t)NROW * 48);
  half_t* pi16   = ah((size_t)NROW * 384);
  half_t* g16    = ah((size_t)NROW * 192);
  float*  ffn    = af((size_t)NROW * 48);

  // ---- weight conversion (f32 -> f16) ----
  long nwc = (long)G4 * KC;
  k_build_wc<<<g256(nwc), 256, 0, stream>>>(intra_wif, intra_whf, wc_if);
  k_build_wc<<<g256(nwc), 256, 0, stream>>>(intra_wib, intra_whb, wc_ib);
  k_build_wc<<<g256(nwc), 256, 0, stream>>>(inter_wif, inter_whf, wc_ef);
  k_build_wc<<<g256(nwc), 256, 0, stream>>>(inter_wib, inter_whb, wc_eb);
  k_f2h<<<g256(KC * HID), 256, 0, stream>>>(il_w,  ilw16,  (long)KC * HID);
  k_f2h<<<g256(KC * HID), 256, 0, stream>>>(itl_w, itlw16, (long)KC * HID);
  k_f2h<<<g256(144 * 48), 256, 0, stream>>>(af_in_w,  afin16,  144 * 48);
  k_f2h<<<g256(48 * 48),  256, 0, stream>>>(af_out_w, afout16, 48 * 48);
  k_f2h<<<g256(144 * 48), 256, 0, stream>>>(at_in_w,  atin16,  144 * 48);
  k_f2h<<<g256(48 * 48),  256, 0, stream>>>(at_out_w, atout16, 48 * 48);
  k_f2h<<<g256(32 * 48),  256, 0, stream>>>(qw, qw16, 32 * 48);
  k_f2h<<<g256(32 * 48),  256, 0, stream>>>(kw, kw16, 32 * 48);
  k_f2h<<<g256(48 * 48),  256, 0, stream>>>(vw, vw16, 48 * 48);
  k_f2h<<<g256(48 * 48),  256, 0, stream>>>(pw, pw16, 48 * 48);
  k_f2h<<<g256(384 * 48), 256, 0, stream>>>(fin_w,  finw16,  384 * 48);
  k_f2h<<<g256(48 * 192), 256, 0, stream>>>(fout_w, foutw16, 48 * 192);

  const float sc12  = 0.28867513459f;            // 1/sqrt(12)
  const float sc520 = 1.0f / sqrtf(520.0f);

  // ================= intra path (over Q) =================
  k_ln_ch<<<g256(NROW), 256, 0, stream>>>(x, intra_ng, intra_nb, seq, seqh, 0);
  gemm(stream, seqh, afin16, af_in_b, qkv, nullptr, NROW, 144, 48, true);
  k_mha<<<g256(NROW), 256, 0, stream>>>(qkv, o16, Qq, sc12);
  gemm(stream, o16, afout16, af_out_b, att, nullptr, NROW, 48, 48, true);
  k_add_ln<<<g256(NROW), 256, 0, stream>>>(att, seq, naf_g, naf_b, seq2);
  k_unfold<<<g256((long)NA * LA * HID), 256, 0, stream>>>(seq2, u16, NA, Qq, LA);
  k_lstm_scan<<<(NA + 15) / 16, 256, 0, stream>>>(u16, wc_if, intra_bif, intra_bhf, rcat, NA, LA, 0, 0);
  k_lstm_scan<<<(NA + 15) / 16, 256, 0, stream>>>(u16, wc_ib, intra_bib, intra_bhb, rcat, NA, LA, 1, HID);
  gemm(stream, rcat, ilw16, nullptr, y2, nullptr, NA * LA, HID, KC, false);
  k_convtr_add<<<g256(Bb * Cc * Tt * Qq), 256, 0, stream>>>(y2, il_b, x, intra, LA, 0);

  // ================= inter path (over T) =================
  k_ln_ch<<<g256(NROW), 256, 0, stream>>>(intra, inter_ng, inter_nb, seq, seqh, 1);
  gemm(stream, seqh, atin16, at_in_b, qkv, nullptr, NROW, 144, 48, true);
  k_mha<<<g256(NROW), 256, 0, stream>>>(qkv, o16, Tt, sc12);
  gemm(stream, o16, atout16, at_out_b, att, nullptr, NROW, 48, 48, true);
  k_add_ln<<<g256(NROW), 256, 0, stream>>>(att, seq, nat_g, nat_b, seq2);
  k_unfold<<<g256((long)NB * LB * HID), 256, 0, stream>>>(seq2, u16, NB, Tt, LB);
  k_lstm_scan<<<(NB + 15) / 16, 256, 0, stream>>>(u16, wc_ef, inter_bif, inter_bhf, rcat, NB, LB, 0, 0);
  k_lstm_scan<<<(NB + 15) / 16, 256, 0, stream>>>(u16, wc_eb, inter_bib, inter_bhb, rcat, NB, LB, 1, HID);
  gemm(stream, rcat, itlw16, nullptr, y2, nullptr, NB * LB, HID, KC, false);
  k_convtr_add<<<g256(Bb * Cc * Tt * Qq), 256, 0, stream>>>(y2, itl_b, intra, inter, LB, 1);

  // ================= E*F head attention over T =================
  k_bt16<<<g256(Bb * Cc * Tt * Qq), 256, 0, stream>>>(inter, bt16);
  gemm(stream, bt16, qw16, nullptr, Qy, nullptr, NROW, 32, 48, true);
  gemm(stream, bt16, kw16, nullptr, Ky, nullptr, NROW, 32, 48, true);
  gemm(stream, bt16, vw16, nullptr, Vy, nullptr, NROW, 48, 48, true);
  k_head_ln<<<4 * Bb * Tt, 128, 0, stream>>>(Qy, qb, qa, qg, qbt, Qf, 8);
  k_head_ln<<<4 * Bb * Tt, 128, 0, stream>>>(Ky, kb, ka, kg, kbt, Kf, 8);
  k_head_ln<<<4 * Bb * Tt, 128, 0, stream>>>(Vy, vb, va, vg, vbt, Vf, 12);
  for (int hb = 0; hb < 8; ++hb)
    gemm(stream, Qf + (size_t)hb * 250 * 520, Kf + (size_t)hb * 250 * 520, nullptr,
         sc + (size_t)hb * 250 * 250, nullptr, 250, 250, 520, true);
  k_softmax<<<g256(8 * 250), 256, 0, stream>>>(sc, pm, 8 * 250, 250, sc520);
  for (int hb = 0; hb < 8; ++hb)
    gemm(stream, pm + (size_t)hb * 250 * 250, Vf + (size_t)hb * 250 * 780, nullptr,
         vo + (size_t)hb * 250 * 780, nullptr, 250, 780, 250, false);
  k_vo16<<<g256(Bb * 48 * Tt * Qq), 256, 0, stream>>>(vo, vo16);
  gemm(stream, vo16, pw16, pb, pj, nullptr, NROW, 48, 48, true);
  k_pj_ln_add<<<Bb * Tt, 256, 0, stream>>>(pj, pa, pg, pbt, inter, outsum, out16);

  // ================= FFN =================
  gemm(stream, out16, finw16, fin_b, nullptr, pi16, NROW, 384, 48, true);
  k_dwglu<<<g256((long)NROW * 192), 256, 0, stream>>>(pi16, fdw_w, fdw_b, g16);
  gemm(stream, g16, foutw16, fout_b, ffn, nullptr, NROW, 48, 192, true);
  k_final<<<g256(Bb * Cc * Tt * Qq), 256, 0, stream>>>(ffn, outsum, (float*)d_out);
}
